// FastLearnedCellX3_84670985273579
// MI455X (gfx1250) — compile-verified, matrix operand order checked by reading, not run on hardware
//
#include <hip/hip_runtime.h>
#include <math.h>

// ---------------------------------------------------------------------------
// Problem constants (from reference): B=4, T=2048 -> N=8192 tokens
// D_in=1024, H=4096, D_out=1024, L=12 experts, top-K=3, tau=1.0, eps=1e-8
// ---------------------------------------------------------------------------
constexpr int N_TOK = 8192;
constexpr int D_IN  = 1024;
constexpr int HID   = 4096;
constexpr int D_OUT = 1024;
constexpr int NEXP  = 12;
constexpr int D_ADDR = 64;

// GEMM tiling: block = 128 tokens x 128 outputs, K-step 32 (WMMA bf16 K)
constexpr int TM = 128;
constexpr int TN = 128;
constexpr int TK = 32;

typedef __attribute__((ext_vector_type(16))) __bf16 v16bf;
typedef __attribute__((ext_vector_type(8)))  __bf16 v8bf;
typedef __attribute__((ext_vector_type(8)))  float  v8f;

union ABFrag { v16bf v; v8bf h[2]; };

__device__ __forceinline__ v8bf cvt8(const float4& f0, const float4& f1) {
    v8bf t;
    t[0] = (__bf16)f0.x; t[1] = (__bf16)f0.y; t[2] = (__bf16)f0.z; t[3] = (__bf16)f0.w;
    t[4] = (__bf16)f1.x; t[5] = (__bf16)f1.y; t[6] = (__bf16)f1.z; t[7] = (__bf16)f1.w;
    return t;
}

// ---------------------------------------------------------------------------
// Kernel 1: routing. One block per token.
// ---------------------------------------------------------------------------
__global__ __launch_bounds__(256)
void route_kernel(const float* __restrict__ x,  const float* __restrict__ Pw,
                  const float* __restrict__ U1, const float* __restrict__ U2,
                  const float* __restrict__ U3,
                  float* __restrict__ dw3,
                  int* __restrict__ cnt1, int* __restrict__ lst1, float* __restrict__ wl1,
                  int* __restrict__ cnt2, int* __restrict__ lst2, float* __restrict__ wl2)
{
    __shared__ __align__(16) float xs[D_IN];
    __shared__ float part[256];
    __shared__ float xaddr[D_ADDR];
    __shared__ float zs[3][NEXP];

    const int n   = blockIdx.x;
    const int tid = threadIdx.x;

    ((float4*)xs)[tid] = ((const float4*)(x + (size_t)n * D_IN))[tid];
    __syncthreads();

    // x_addr: j = tid/4, quarter = tid%4 covers 256 elements
    {
        const int j = tid >> 2, q = tid & 3;
        const float4* pr = (const float4*)(Pw + (size_t)j * D_IN + q * 256);
        const float4* xr = (const float4*)(xs + q * 256);
        float s = 0.f;
        #pragma unroll 4
        for (int k = 0; k < 64; ++k) {
            float4 a = pr[k], b = xr[k];
            s += a.x * b.x + a.y * b.y + a.z * b.z + a.w * b.w;
        }
        part[tid] = s;
    }
    __syncthreads();
    if (tid < D_ADDR)
        xaddr[tid] = part[tid * 4] + part[tid * 4 + 1] + part[tid * 4 + 2] + part[tid * 4 + 3];
    __syncthreads();

    if (tid < 3 * NEXP) {
        const int r = tid / NEXP, e = tid % NEXP;
        const float* U = (r == 0) ? U1 : (r == 1) ? U2 : U3;
        const float* u = U + e * D_ADDR;
        float z = 0.f;
        #pragma unroll 8
        for (int k = 0; k < D_ADDR; ++k) z += xaddr[k] * u[k];
        zs[r][e] = z;
    }
    __syncthreads();

    if (tid < 3) {
        const int r = tid;
        float v[NEXP];
        #pragma unroll
        for (int e = 0; e < NEXP; ++e) v[e] = zs[r][e];
        int   ei[3];
        float ev[3];
        #pragma unroll
        for (int p = 0; p < 3; ++p) {
            int   bi = 0;
            float bv = -3.4e38f;
            #pragma unroll
            for (int e = 0; e < NEXP; ++e) {
                if (v[e] > bv) { bv = v[e]; bi = e; }
            }
            ei[p] = bi; ev[p] = bv; v[bi] = -3.4e38f;
        }
        const float invtau = 1.0f / (1.0f + 1e-8f);
        float mx = ev[0];
        float e0 = __expf((ev[0] - mx) * invtau);
        float e1 = __expf((ev[1] - mx) * invtau);
        float e2 = __expf((ev[2] - mx) * invtau);
        float inv = 1.0f / (e0 + e1 + e2);
        float w[3] = { e0 * inv, e1 * inv, e2 * inv };

        if (r == 2) {
            float row[NEXP];
            #pragma unroll
            for (int e = 0; e < NEXP; ++e) row[e] = 0.f;
            row[ei[0]] = w[0]; row[ei[1]] = w[1]; row[ei[2]] = w[2];
            #pragma unroll
            for (int e = 0; e < NEXP; ++e) dw3[(size_t)n * NEXP + e] = row[e];
        } else {
            int*   cnt = (r == 0) ? cnt1 : cnt2;
            int*   lst = (r == 0) ? lst1 : lst2;
            float* wl  = (r == 0) ? wl1  : wl2;
            #pragma unroll
            for (int p = 0; p < 3; ++p) {
                int pos = atomicAdd(&cnt[ei[p]], 1);
                lst[ei[p] * N_TOK + pos] = n;
                wl [ei[p] * N_TOK + pos] = w[p];
            }
        }
    }
}

// ---------------------------------------------------------------------------
// Kernel 2: y = dw3 @ b2 (initializes d_out accumulator). One block per token.
// ---------------------------------------------------------------------------
__global__ __launch_bounds__(256)
void bias_kernel(const float* __restrict__ dw3, const float* __restrict__ b2,
                 float* __restrict__ y)
{
    __shared__ float w[NEXP];
    const int n = blockIdx.x, tid = threadIdx.x;
    if (tid < NEXP) w[tid] = dw3[(size_t)n * NEXP + tid];
    __syncthreads();
    for (int d = tid; d < D_OUT; d += 256) {
        float acc = 0.f;
        #pragma unroll
        for (int e = 0; e < NEXP; ++e) acc += w[e] * b2[e * D_OUT + d];
        y[(size_t)n * D_OUT + d] = acc;
    }
}

// ---------------------------------------------------------------------------
// Kernel 3: grouped expert GEMM (both layers), double-buffered + pipelined.
//   Out[tok, n0:n0+TN] += rowW * (A[tok, :] @ W[e, n0:n0+TN, :]^T)
//
// 8 waves: 4 over M (32-row strips) x 2 over N (64-col strips).
// Per K-step each wave runs 8 x v_wmma_f32_16x16x32_bf16 (2 A-frags sharing
// each of 4 B-frags). Next K-tile's global_load_b128s are issued before the
// compute phase so HBM latency hides under the WMMAs. One barrier/iteration:
// barrier(i-1) transitively orders compute(i-2) (last reader of the buffer
// being overwritten during stage(i)).
//
// WMMA 16-bit operand layout (ISA 7.12.2): lane half h = lane/16,
// frag elems 0..7 = K in [h*8, h*8+8), elems 8..15 = K in [16+h*8, +8)
// -> two 16B LDS loads per fragment.
// ---------------------------------------------------------------------------
__global__ __launch_bounds__(256)
void moe_gemm(const float* __restrict__ A, int Dk,
              const float* __restrict__ W, int Dn,
              const int* __restrict__ cnt,
              const int* __restrict__ lst, const float* __restrict__ wl,
              float* __restrict__ Out)
{
    const int e  = blockIdx.z;
    const int m0 = blockIdx.y * TM;
    const int n0 = blockIdx.x * TN;
    const int c  = cnt[e];
    if (m0 >= c) return;

    __shared__ __align__(16) __bf16 As[2][TM][TK];   // 2 x 8 KB
    __shared__ __align__(16) __bf16 Bs[2][TN][TK];   // 2 x 8 KB
    __shared__ int   rIdx[TM];
    __shared__ float rW[TM];

    const int tid = threadIdx.x;
    if (tid < TM) {
        const int m = m0 + tid;
        if (m < c) { rIdx[tid] = lst[e * N_TOK + m]; rW[tid] = wl[e * N_TOK + m]; }
        else       { rIdx[tid] = -1;                 rW[tid] = 0.f; }
    }
    __syncthreads();

    // staging assignment (both A and B): row = tid/2, 16 cols at (tid%2)*16
    const int sRow = tid >> 1;
    const int sCol = (tid & 1) * 16;
    const int aRowG = rIdx[sRow];
    const float* aPtr = (aRowG >= 0) ? (A + (size_t)aRowG * Dk + sCol) : nullptr;
    const float* bPtr = W + (size_t)e * Dn * Dk + (size_t)(n0 + sRow) * Dk + sCol;

    const int wave = tid >> 5;
    const int lane = tid & 31;
    const int wm   = wave & 3;     // M strip: rows [wm*32, wm*32+32)
    const int wn   = wave >> 2;    // N strip: cols [wn*64, wn*64+64)
    const int hh   = lane >> 4;
    const int l16  = lane & 15;

    v8f acc[2][4];
    #pragma unroll
    for (int a = 0; a < 2; ++a)
        #pragma unroll
        for (int s = 0; s < 4; ++s)
            #pragma unroll
            for (int r = 0; r < 8; ++r) acc[a][s][r] = 0.f;

    float4 ar[4], br[4];

    // prologue: load K-tile 0
    {
        if (aPtr) {
            const float4* p = (const float4*)aPtr;
            ar[0] = p[0]; ar[1] = p[1]; ar[2] = p[2]; ar[3] = p[3];
        } else {
            ar[0] = ar[1] = ar[2] = ar[3] = make_float4(0.f, 0.f, 0.f, 0.f);
        }
        const float4* q = (const float4*)bPtr;
        br[0] = q[0]; br[1] = q[1]; br[2] = q[2]; br[3] = q[3];
    }

    for (int kk = 0;; kk += TK) {
        const int buf = (kk >> 5) & 1;

        // stage current K-tile into LDS (f32 -> bf16)
        *(v8bf*)&As[buf][sRow][sCol]     = cvt8(ar[0], ar[1]);
        *(v8bf*)&As[buf][sRow][sCol + 8] = cvt8(ar[2], ar[3]);
        *(v8bf*)&Bs[buf][sRow][sCol]     = cvt8(br[0], br[1]);
        *(v8bf*)&Bs[buf][sRow][sCol + 8] = cvt8(br[2], br[3]);

        // issue next K-tile's global loads (hidden under this tile's WMMAs)
        const int kn = kk + TK;
        const bool more = kn < Dk;
        if (more) {
            if (aPtr) {
                const float4* p = (const float4*)(aPtr + kn);
                ar[0] = p[0]; ar[1] = p[1]; ar[2] = p[2]; ar[3] = p[3];
            } else {
                ar[0] = ar[1] = ar[2] = ar[3] = make_float4(0.f, 0.f, 0.f, 0.f);
            }
            const float4* q = (const float4*)(bPtr + kn);
            br[0] = q[0]; br[1] = q[1]; br[2] = q[2]; br[3] = q[3];
        }

        __syncthreads();

        // compute: 2 A-frags x 4 B-frags = 8 WMMAs
        ABFrag a0, a1;
        a0.h[0] = *(const v8bf*)&As[buf][wm * 32 + l16][hh * 8];
        a0.h[1] = *(const v8bf*)&As[buf][wm * 32 + l16][16 + hh * 8];
        a1.h[0] = *(const v8bf*)&As[buf][wm * 32 + 16 + l16][hh * 8];
        a1.h[1] = *(const v8bf*)&As[buf][wm * 32 + 16 + l16][16 + hh * 8];
        #pragma unroll
        for (int s = 0; s < 4; ++s) {
            const int nrow = wn * 64 + s * 16 + l16;
            ABFrag b;
            b.h[0] = *(const v8bf*)&Bs[buf][nrow][hh * 8];
            b.h[1] = *(const v8bf*)&Bs[buf][nrow][16 + hh * 8];
            acc[0][s] = __builtin_amdgcn_wmma_f32_16x16x32_bf16(
                false, a0.v, false, b.v, (short)0, acc[0][s], false, false);
            acc[1][s] = __builtin_amdgcn_wmma_f32_16x16x32_bf16(
                false, a1.v, false, b.v, (short)0, acc[1][s], false, false);
        }

        if (!more) break;
    }

    // epilogue: scale by routing weight, atomic-accumulate into Out
    // C layout: VGPR r -> M = (lane/16)*8 + r, N = lane%16
    #pragma unroll
    for (int asub = 0; asub < 2; ++asub) {
        #pragma unroll
        for (int s = 0; s < 4; ++s) {
            const int gn = n0 + wn * 64 + s * 16 + l16;
            #pragma unroll
            for (int r = 0; r < 8; ++r) {
                const int ml   = wm * 32 + asub * 16 + hh * 8 + r;
                const int grow = rIdx[ml];
                if (grow >= 0) {
                    atomicAdd(&Out[(size_t)grow * Dn + gn], acc[asub][s][r] * rW[ml]);
                }
            }
        }
    }
}

// ---------------------------------------------------------------------------
// Kernel 4: exact GELU in place over h
// ---------------------------------------------------------------------------
__global__ __launch_bounds__(256)
void gelu_kernel(float* __restrict__ h, long long n4)
{
    const long long stride = (long long)gridDim.x * blockDim.x;
    for (long long i = (long long)blockIdx.x * blockDim.x + threadIdx.x; i < n4; i += stride) {
        float4 v = ((float4*)h)[i];
        v.x = 0.5f * v.x * (1.0f + erff(v.x * 0.70710678118654752f));
        v.y = 0.5f * v.y * (1.0f + erff(v.y * 0.70710678118654752f));
        v.z = 0.5f * v.z * (1.0f + erff(v.z * 0.70710678118654752f));
        v.w = 0.5f * v.w * (1.0f + erff(v.w * 0.70710678118654752f));
        ((float4*)h)[i] = v;
    }
}

// ---------------------------------------------------------------------------
// Launch
// ---------------------------------------------------------------------------
extern "C" void kernel_launch(void* const* d_in, const int* in_sizes, int n_in,
                              void* d_out, int out_size, void* d_ws, size_t ws_size,
                              hipStream_t stream)
{
    const float* x  = (const float*)d_in[0];   // [8192,1024]
    const float* Pw = (const float*)d_in[1];   // [64,1024]
    const float* U1 = (const float*)d_in[2];   // [12,64]
    const float* U2 = (const float*)d_in[3];
    const float* U3 = (const float*)d_in[4];
    const float* W1 = (const float*)d_in[5];   // [12,4096,1024]
    const float* W2 = (const float*)d_in[6];   // [12,1024,4096]
    const float* b2 = (const float*)d_in[7];   // [12,1024]
    float* y = (float*)d_out;                  // [8192,1024]

    // workspace layout
    float* h    = (float*)d_ws;                           // 8192*4096 f32
    float* dw3  = h + (size_t)N_TOK * HID;                // 8192*12
    int*   cnt1 = (int*)(dw3 + (size_t)N_TOK * NEXP);     // 16 ints
    int*   cnt2 = cnt1 + 16;                              // 16 ints
    int*   lst1 = cnt2 + 16;                              // 12*8192
    int*   lst2 = lst1 + NEXP * N_TOK;
    float* wl1  = (float*)(lst2 + NEXP * N_TOK);
    float* wl2  = wl1 + NEXP * N_TOK;

    hipMemsetAsync(h, 0, (size_t)N_TOK * HID * sizeof(float), stream);
    hipMemsetAsync(cnt1, 0, 32 * sizeof(int), stream);

    // 1) routing
    route_kernel<<<N_TOK, 256, 0, stream>>>(x, Pw, U1, U2, U3,
                                            dw3, cnt1, lst1, wl1, cnt2, lst2, wl2);
    // 2) y = dw3 @ b2 (init output accumulator)
    bias_kernel<<<N_TOK, 256, 0, stream>>>(dw3, b2, y);
    // 3) layer 1 grouped GEMM: h += dw1 .* (x @ W1^T)
    moe_gemm<<<dim3(HID / TN, N_TOK / TM, NEXP), 256, 0, stream>>>(
        x, D_IN, W1, HID, cnt1, lst1, wl1, h);
    // 4) exact GELU
    gelu_kernel<<<4096, 256, 0, stream>>>(h, (long long)N_TOK * HID / 4);
    // 5) layer 2 grouped GEMM: y += dw2 .* (h @ W2^T)
    moe_gemm<<<dim3(D_OUT / TN, N_TOK / TM, NEXP), 256, 0, stream>>>(
        h, HID, W2, D_OUT, cnt2, lst2, wl2, y);
}
